// SimpleGATNet_47708496724079
// MI455X (gfx1250) — compile-verified, matrix operand order checked by reading, not run on hardware
//
#include <hip/hip_runtime.h>
#include <hip/hip_bf16.h>
#include <math.h>

#define GAT_N_IN   128
#define GAT_HID    32
#define GAT_HEADS  4
#define GAT_OUT    16
#define GAT_SLOPE  0.2f

typedef __attribute__((ext_vector_type(16))) __bf16        v16bf;
typedef __attribute__((ext_vector_type(8)))  float         v8f;
typedef __attribute__((ext_vector_type(4)))  unsigned int  u32x4;

// fragment = two contiguous 16-byte runs of bf16, loaded as 2x b128
union Frag16 {
    u32x4 u[2];
    v16bf v;
};

// ---------------------------------------------------------------------------
// GEMM: C[M, NTILES*16] = A[M, KC] @ Wt^T, all-bf16 operands, f32 accumulate.
//   A  : [M, KC]     bf16 row-major (pre-converted)
//   Wt : [Ncols, KC] bf16 (pre-converted, transposed W) -> B fragment loads are
//        two contiguous b128s per K-chunk, exactly like A.
// One wave computes a 16-row x (NTILES*16)-col strip; A fragment loaded once
// per 32-wide K chunk and reused across NTILES WMMAs. K loop kept rolled to
// bound VGPR pressure (acc 64 + frags ~ 150 VGPRs, no spills).
// Fragment striping (16-bit): lane<16 -> K {0..7,16..23}, lane>=16 -> {8..15,24..31}
// C/D: VGPR r holds row (r + 8*half), column = lane&15.
// ---------------------------------------------------------------------------
template <int KC, int NTILES>
__global__ void gat_gemm_wmma(const __bf16* __restrict__ A,
                              const __bf16* __restrict__ Wt,
                              float* __restrict__ C,
                              int Ncols) {
    const int row0  = blockIdx.x * 16;
    const int lane  = threadIdx.x & 31;
    const int half  = lane >> 4;      // 0: lanes 0-15, 1: lanes 16-31
    const int idx16 = lane & 15;

    v8f acc[NTILES];
#pragma unroll
    for (int t = 0; t < NTILES; ++t) acc[t] = (v8f){};

    const __bf16* arow = A + (size_t)(row0 + idx16) * KC + half * 8;

#pragma unroll 1
    for (int kc = 0; kc < KC; kc += 32) {
        Frag16 af;
        af.u[0] = *(const u32x4*)(arow + kc);
        af.u[1] = *(const u32x4*)(arow + kc + 16);

#pragma unroll
        for (int t = 0; t < NTILES; ++t) {
            const __bf16* bcol = Wt + (size_t)(t * 16 + idx16) * KC + half * 8;
            Frag16 bf;
            bf.u[0] = *(const u32x4*)(bcol + kc);
            bf.u[1] = *(const u32x4*)(bcol + kc + 16);
            acc[t] = __builtin_amdgcn_wmma_f32_16x16x32_bf16(
                /*neg_a=*/false, af.v, /*neg_b=*/false, bf.v,
                /*c_mod=*/(short)0, acc[t], /*reuse_a=*/false, /*reuse_b=*/false);
        }
    }

#pragma unroll
    for (int t = 0; t < NTILES; ++t) {
#pragma unroll
        for (int r = 0; r < 8; ++r) {
            const int m = r + half * 8;
            C[(size_t)(row0 + m) * Ncols + t * 16 + idx16] = acc[t][r];
        }
    }
}

// elementwise f32 -> bf16 (streaming pass; HBM-bandwidth bound and tiny)
__global__ void f32_to_bf16(const float* __restrict__ in, __bf16* __restrict__ out,
                            int count) {
    const int i = blockIdx.x * blockDim.x + threadIdx.x;
    if (i < count) out[i] = (__bf16)in[i];
}

// W[K, Ncols] f32 -> Wt[Ncols, K] bf16 (transpose + convert; W is tiny)
__global__ void f32_to_bf16_T(const float* __restrict__ in, __bf16* __restrict__ out,
                              int K, int Ncols) {
    const int i = blockIdx.x * blockDim.x + threadIdx.x;
    if (i >= K * Ncols) return;
    const int nn = i / K;
    const int k  = i % K;
    out[i] = (__bf16)in[(size_t)k * Ncols + nn];
}

// ---------------------------------------------------------------------------
// Per-node attention logit halves: a_s[n,h] = <h[n,h,:], att_src[h,:]>
// ---------------------------------------------------------------------------
__global__ void gat_logits(const float* __restrict__ h,
                           const float* __restrict__ att_src,
                           const float* __restrict__ att_dst,
                           float* __restrict__ a_s, float* __restrict__ a_d,
                           int n, int heads, int hid) {
    const int idx = blockIdx.x * blockDim.x + threadIdx.x;
    if (idx >= n * heads) return;
    const int hd = idx % heads;
    const float* hp = h + (size_t)idx * hid;
    float s = 0.f, d = 0.f;
    for (int c = 0; c < hid; ++c) {
        const float v = hp[c];
        s += v * att_src[hd * hid + c];
        d += v * att_dst[hd * hid + c];
    }
    a_s[idx] = s;
    a_d[idx] = d;
}

__global__ void gat_init_maxden(float* __restrict__ m, float* __restrict__ den, int count) {
    const int i = blockIdx.x * blockDim.x + threadIdx.x;
    if (i < count) { m[i] = -__builtin_inff(); den[i] = 0.f; }
}

__device__ __forceinline__ void atomicMaxF(float* addr, float v) {
    if (v >= 0.f) atomicMax((int*)addr, __float_as_int(v));
    else          atomicMin((unsigned int*)addr, __float_as_uint(v));
}

__device__ __forceinline__ float leaky(float e) {
    return e > 0.f ? e : GAT_SLOPE * e;
}

// edges i in [0,E) are real edges; i in [E, E+n) are self loops (PyG default)
__global__ void gat_edge_max(const long long* __restrict__ ei,
                             const float* __restrict__ a_s, const float* __restrict__ a_d,
                             float* __restrict__ m, int E, int n, int heads) {
    const int i = blockIdx.x * blockDim.x + threadIdx.x;
    if (i >= E + n) return;
    const int src = (i < E) ? (int)ei[i]     : (i - E);
    const int dst = (i < E) ? (int)ei[E + i] : (i - E);
    for (int hh = 0; hh < heads; ++hh) {
        const float e = leaky(a_s[src * heads + hh] + a_d[dst * heads + hh]);
        atomicMaxF(&m[dst * heads + hh], e);
    }
}

__global__ void gat_edge_sum(const long long* __restrict__ ei,
                             const float* __restrict__ a_s, const float* __restrict__ a_d,
                             const float* __restrict__ m, float* __restrict__ den,
                             int E, int n, int heads) {
    const int i = blockIdx.x * blockDim.x + threadIdx.x;
    if (i >= E + n) return;
    const int src = (i < E) ? (int)ei[i]     : (i - E);
    const int dst = (i < E) ? (int)ei[E + i] : (i - E);
    for (int hh = 0; hh < heads; ++hh) {
        const float e = leaky(a_s[src * heads + hh] + a_d[dst * heads + hh]);
        atomicAdd(&den[dst * heads + hh], __expf(e - m[dst * heads + hh]));
    }
}

// One wave per edge; lanes sweep the C = heads*hid channels.
__global__ void gat_edge_aggr(const long long* __restrict__ ei,
                              const float* __restrict__ a_s, const float* __restrict__ a_d,
                              const float* __restrict__ m, const float* __restrict__ den,
                              const float* __restrict__ h, float* __restrict__ out,
                              int E, int n, int heads, int hid) {
    const int wid  = (blockIdx.x * blockDim.x + threadIdx.x) >> 5;
    const int lane = threadIdx.x & 31;
    if (wid >= E + n) return;
    const int src = (wid < E) ? (int)ei[wid]     : (wid - E);
    const int dst = (wid < E) ? (int)ei[E + wid] : (wid - E);
    const int C = heads * hid;
    for (int base = lane; base < C; base += 32) {
        const int hh = base / hid;
        const float e = leaky(a_s[src * heads + hh] + a_d[dst * heads + hh]);
        const float alpha = __expf(e - m[dst * heads + hh]) /
                            (den[dst * heads + hh] + 1e-16f);
        atomicAdd(&out[(size_t)dst * C + base], alpha * h[(size_t)src * C + base]);
    }
}

// x1 = ELU(agg + b)   (in place)
__global__ void gat_bias_elu(float* __restrict__ x, const float* __restrict__ b,
                             int total, int C) {
    const int i = blockIdx.x * blockDim.x + threadIdx.x;
    if (i >= total) return;
    const float v = x[i] + b[i % C];
    x[i] = v > 0.f ? v : (__expf(v) - 1.f);
}

// out = agg + b   (in place, no activation on final layer)
__global__ void gat_bias(float* __restrict__ x, const float* __restrict__ b,
                         int total, int C) {
    const int i = blockIdx.x * blockDim.x + threadIdx.x;
    if (i >= total) return;
    x[i] += b[i % C];
}

extern "C" void kernel_launch(void* const* d_in, const int* in_sizes, int n_in,
                              void* d_out, int out_size, void* d_ws, size_t ws_size,
                              hipStream_t stream) {
    const float* x    = (const float*)d_in[0];
    const float* W1   = (const float*)d_in[1];
    const float* as1w = (const float*)d_in[2];
    const float* ad1w = (const float*)d_in[3];
    const float* b1   = (const float*)d_in[4];
    const float* W2   = (const float*)d_in[5];
    const float* as2w = (const float*)d_in[6];
    const float* ad2w = (const float*)d_in[7];
    const float* b2   = (const float*)d_in[8];
    const long long* ei = (const long long*)d_in[9];

    const int n  = in_sizes[0] / GAT_N_IN;         // 100000 (divisible by 16)
    const int E  = in_sizes[9] / 2;                // 1600000
    const int C1 = GAT_HEADS * GAT_HID;            // 128
    const int total_edges = E + n;

    // ---- workspace carve-up ----
    float* ws  = (float*)d_ws;
    float* h1  = ws;  ws += (size_t)n * C1;        // layer1 features   [N,128] f32
    float* x1  = ws;  ws += (size_t)n * C1;        // agg1 -> ELU -> x1 [N,128] f32
    float* aS1 = ws;  ws += (size_t)n * GAT_HEADS;
    float* aD1 = ws;  ws += (size_t)n * GAT_HEADS;
    float* m1  = ws;  ws += (size_t)n * GAT_HEADS;
    float* dn1 = ws;  ws += (size_t)n * GAT_HEADS;
    float* h2  = ws;  ws += (size_t)n * GAT_OUT;   // layer2 features   [N,16] f32
    float* aS2 = ws;  ws += (size_t)n;
    float* aD2 = ws;  ws += (size_t)n;
    float* m2  = ws;  ws += (size_t)n;
    float* dn2 = ws;  ws += (size_t)n;
    __bf16* abuf = (__bf16*)ws;                    // bf16 activations [N,128]
    ws += (size_t)n * C1 / 2;                      //   (reused: x then x1)
    __bf16* W1t  = (__bf16*)ws;                    // bf16 W1^T [128,128]
    ws += (size_t)GAT_N_IN * C1 / 2;
    __bf16* W2t  = (__bf16*)ws;                    // bf16 W2^T [16,128]
    ws += (size_t)C1 * GAT_OUT / 2;
    float* out = (float*)d_out;                    // agg2 accumulates here [N,16]

    const int B = 256;
    const int gNH1 = (n * GAT_HEADS + B - 1) / B;
    const int gE   = (total_edges + B - 1) / B;
    const int gEw  = ((total_edges * 32) + B - 1) / B;   // wave-per-edge
    const int gNC1 = (n * C1 + B - 1) / B;
    const int gNC2 = (n * GAT_OUT + B - 1) / B;

    // ---- weight preconversion (tiny) ----
    f32_to_bf16_T<<<(GAT_N_IN * C1 + B - 1) / B, B, 0, stream>>>(W1, W1t, GAT_N_IN, C1);
    f32_to_bf16_T<<<(C1 * GAT_OUT + B - 1) / B, B, 0, stream>>>(W2, W2t, C1, GAT_OUT);

    // ================= Layer 1 =================
    f32_to_bf16<<<gNC1, B, 0, stream>>>(x, abuf, n * C1);
    // h1 = x @ W1 : one wave per 16x128 strip; per K-chunk: 2+16 b128 loads, 8 WMMAs
    gat_gemm_wmma<GAT_N_IN, 8><<<n / 16, 32, 0, stream>>>(abuf, W1t, h1, C1);
    gat_logits<<<gNH1, B, 0, stream>>>(h1, as1w, ad1w, aS1, aD1, n, GAT_HEADS, GAT_HID);
    gat_init_maxden<<<gNH1, B, 0, stream>>>(m1, dn1, n * GAT_HEADS);
    hipMemsetAsync(x1, 0, (size_t)n * C1 * sizeof(float), stream);

    gat_edge_max <<<gE,  B, 0, stream>>>(ei, aS1, aD1, m1, E, n, GAT_HEADS);
    gat_edge_sum <<<gE,  B, 0, stream>>>(ei, aS1, aD1, m1, dn1, E, n, GAT_HEADS);
    gat_edge_aggr<<<gEw, B, 0, stream>>>(ei, aS1, aD1, m1, dn1, h1, x1,
                                         E, n, GAT_HEADS, GAT_HID);
    gat_bias_elu<<<gNC1, B, 0, stream>>>(x1, b1, n * C1, C1);

    // ================= Layer 2 =================
    f32_to_bf16<<<gNC1, B, 0, stream>>>(x1, abuf, n * C1);
    // h2 = x1 @ W2 : one wave per 16x16 strip
    gat_gemm_wmma<C1, 1><<<n / 16, 32, 0, stream>>>(abuf, W2t, h2, GAT_OUT);
    gat_logits<<<(n + B - 1) / B, B, 0, stream>>>(h2, as2w, ad2w, aS2, aD2, n, 1, GAT_OUT);
    gat_init_maxden<<<(n + B - 1) / B, B, 0, stream>>>(m2, dn2, n);
    hipMemsetAsync(out, 0, (size_t)n * GAT_OUT * sizeof(float), stream);

    gat_edge_max <<<gE,  B, 0, stream>>>(ei, aS2, aD2, m2, E, n, 1);
    gat_edge_sum <<<gE,  B, 0, stream>>>(ei, aS2, aD2, m2, dn2, E, n, 1);
    gat_edge_aggr<<<gEw, B, 0, stream>>>(ei, aS2, aD2, m2, dn2, h2, out,
                                         E, n, 1, GAT_OUT);
    gat_bias<<<gNC2, B, 0, stream>>>(out, b2, n * GAT_OUT, GAT_OUT);
}
